// Spatial2DAttention_77799037599847
// MI455X (gfx1250) — compile-verified, matrix operand order checked by reading, not run on hardware
//
#include <hip/hip_runtime.h>
#include <hip/hip_bf16.h>
#include <stdint.h>

// ---------------------------------------------------------------------------
// Spatial2DAttention fused pipeline for MI455X (gfx1250, wave32, WMMA).
//   B=16, H=W=30, S=900, d_model=512, heads=8, d_k=64.
// Precision: bf16 WMMA operands (fp32-range-safe), f32 accumulation.
// Attention computed fully TRANSPOSED (S^T = K@Q^T, ctx^T = V^T@P^T) so each
// lane owns one query column: online-softmax stats need only ONE cross-lane
// shuffle per block, and the key-block loop is peeled so the 28 full blocks
// carry no edge masking / exec-mask branches at all.
// ---------------------------------------------------------------------------

#define D_MODEL 512
#define M_ROWS  14400           // B * S
#define S_LEN   900
#define NHEADS  8
#define DK      64

typedef __attribute__((ext_vector_type(16))) __bf16 v16bf;
typedef __attribute__((ext_vector_type(8)))  float  v8f;

union FragB16 { v16bf v; uint32_t u[8]; };
union FragF32 { v8f  v; float    f[8]; };

__device__ __forceinline__ uint16_t f32_to_bf16(float f) {
    __bf16 h = (__bf16)f;                       // native v_cvt (RNE) on gfx1250
    return __builtin_bit_cast(uint16_t, h);
}
__device__ __forceinline__ uint32_t pack_bf16(float lo, float hi) {
    return (uint32_t)f32_to_bf16(lo) | ((uint32_t)f32_to_bf16(hi) << 16);
}
// A/B fragment dword j -> starting K index of the 2 packed bf16 (16x16x32 layout):
// lanes 0-15 hold K {0..7, 16..23}, lanes 16-31 hold K {8..15, 24..31}.
__device__ __forceinline__ int kmap(int j, int half) {
    return (j < 4 ? 2 * j : 16 + 2 * (j - 4)) + half * 8;
}

enum { EPI_HEADS = 0, EPI_VT = 1, EPI_RESID = 2 };

// ---------------------------------------------------------------------------
// Generic M=14400 x N=512 x K=512 bf16 WMMA GEMM: out = A @ W + bias (+x).
// Block tile 64x64, 4 waves of 32x32 (2x2 accumulators), K-step 32.
// A (x / ctx, 29.5 MB) and W (1 MB) stay resident in the 192 MB L2, so the
// 8x re-read of A across N-blocks costs L2 bandwidth only.
// ---------------------------------------------------------------------------
template <int EPI>
__global__ __launch_bounds__(128) void gemm512(
    const float* __restrict__ A, const float* __restrict__ W,
    const float* __restrict__ bias, const float* __restrict__ xres,
    void* __restrict__ outp) {
    __shared__ uint16_t lds_a[64 * 32];   // [row][k] bf16
    __shared__ uint16_t lds_bt[64 * 32];  // [n][k]  bf16 (transposed weights)

    const int tid  = threadIdx.x;
    const int wave = tid >> 5, lane = tid & 31;
    const int half = lane >> 4, l16 = lane & 15;
    const int mBase = blockIdx.x * 64;
    const int nBase = blockIdx.y * 64;
    const int wr = (wave >> 1) * 32, wc = (wave & 1) * 32;

    FragF32 acc[2][2];
#pragma unroll
    for (int a = 0; a < 2; ++a)
#pragma unroll
        for (int b = 0; b < 2; ++b)
#pragma unroll
            for (int r = 0; r < 8; ++r) acc[a][b].f[r] = 0.0f;

    for (int k0 = 0; k0 < D_MODEL; k0 += 32) {
        __syncthreads();
        {   // stage A tile: 64 rows x 32 k, f32 -> bf16
            const int row = tid >> 1, col = (tid & 1) * 16;
            const float* src = A + (size_t)(mBase + row) * D_MODEL + k0 + col;
            uint32_t* dst = (uint32_t*)&lds_a[row * 32 + col];
#pragma unroll
            for (int i = 0; i < 4; ++i) {
                float4 f = ((const float4*)src)[i];
                dst[i * 2 + 0] = pack_bf16(f.x, f.y);
                dst[i * 2 + 1] = pack_bf16(f.z, f.w);
            }
        }
        {   // stage W tile transposed: k0..k0+31 rows x 64 cols -> lds_bt[n][k]
            const int kk = tid >> 2, nn = (tid & 3) * 16;
            const float* src = W + (size_t)(k0 + kk) * D_MODEL + nBase + nn;
#pragma unroll
            for (int i = 0; i < 4; ++i) {
                float4 f = ((const float4*)src)[i];
                lds_bt[(nn + i * 4 + 0) * 32 + kk] = f32_to_bf16(f.x);
                lds_bt[(nn + i * 4 + 1) * 32 + kk] = f32_to_bf16(f.y);
                lds_bt[(nn + i * 4 + 2) * 32 + kk] = f32_to_bf16(f.z);
                lds_bt[(nn + i * 4 + 3) * 32 + kk] = f32_to_bf16(f.w);
            }
        }
        __syncthreads();

        FragB16 af[2], bf[2];
#pragma unroll
        for (int rg = 0; rg < 2; ++rg) {
            const int row = wr + rg * 16 + l16;
#pragma unroll
            for (int j = 0; j < 8; ++j)
                af[rg].u[j] = *(const uint32_t*)&lds_a[row * 32 + kmap(j, half)];
        }
#pragma unroll
        for (int cg = 0; cg < 2; ++cg) {
            const int col = wc + cg * 16 + l16;
#pragma unroll
            for (int j = 0; j < 8; ++j)
                bf[cg].u[j] = *(const uint32_t*)&lds_bt[col * 32 + kmap(j, half)];
        }
#pragma unroll
        for (int rg = 0; rg < 2; ++rg)
#pragma unroll
            for (int cg = 0; cg < 2; ++cg)
                acc[rg][cg].v = __builtin_amdgcn_wmma_f32_16x16x32_bf16(
                    false, af[rg].v, false, bf[cg].v, (short)0, acc[rg][cg].v,
                    false, false);
    }

    // epilogue: C layout is VGPR r -> row r (lanes 0-15) / row r+8 (lanes 16-31)
#pragma unroll
    for (int rg = 0; rg < 2; ++rg)
#pragma unroll
        for (int cg = 0; cg < 2; ++cg) {
            const int n = nBase + wc + cg * 16 + l16;
            const float bn = bias[n];
#pragma unroll
            for (int r = 0; r < 8; ++r) {
                const int m = mBase + wr + rg * 16 + r + 8 * half;
                float val = acc[rg][cg].f[r] + bn;
                if (EPI == EPI_HEADS) {
                    const int b = m / S_LEN, s = m % S_LEN, h = n >> 6, dk = n & 63;
                    ((uint16_t*)outp)[(((size_t)(b * NHEADS + h) * S_LEN) + s) * DK + dk] =
                        f32_to_bf16(val);
                } else if (EPI == EPI_VT) {
                    const int b = m / S_LEN, s = m % S_LEN, h = n >> 6, dk = n & 63;
                    ((uint16_t*)outp)[((size_t)(b * NHEADS + h) * DK + dk) * S_LEN + s] =
                        f32_to_bf16(val);
                } else {  // EPI_RESID: out = A@W + bias + x
                    ((float*)outp)[(size_t)m * D_MODEL + n] =
                        val + xres[(size_t)m * D_MODEL + n];
                }
            }
        }
}

// ---------------------------------------------------------------------------
// One 32-key block of transposed flash attention. EDGE=true only for the
// final block (kb=896): all bounds guards compile away in the 28 hot blocks.
// V^T fragments are issued FIRST so their memory latency is hidden behind the
// score WMMAs and the softmax VALU work.
// ---------------------------------------------------------------------------
template <bool EDGE>
__device__ __forceinline__ void attn_block(
    const int kb, const uint16_t* __restrict__ Kb,
    const uint16_t* __restrict__ Vb, const FragB16 (&bq)[2],
    FragF32 (&cacc)[4], float& m_run, float& l_run, const int hqv,
    const int wqv, const int half, const int l16,
    uint16_t* __restrict__ pl, const float* __restrict__ s_bias) {
    // ---- issue V^T A-fragment loads early (consumed after softmax) ----
    FragB16 av[4];
#pragma unroll
    for (int cg = 0; cg < 4; ++cg) {
        const int d = cg * 16 + l16;                  // A row = d
        const uint16_t* vrow = &Vb[(size_t)d * S_LEN + kb];
#pragma unroll
        for (int j = 0; j < 8; ++j) {
            const int ko = kmap(j, half);
            if (EDGE) {
                const int key0 = kb + ko;
                uint32_t u;
                if (key0 + 1 < S_LEN)  u = *(const uint32_t*)&vrow[ko];
                else if (key0 < S_LEN) u = (uint32_t)vrow[ko];
                else                   u = 0u;        // keep 0*P exact (no NaN)
                av[cg].u[j] = u;
            } else {
                av[cg].u[j] = *(const uint32_t*)&vrow[ko];
            }
        }
    }

    // ---- S^T: two 16-key tiles, accumulate over d (4 WMMAs) ----
    FragF32 sc[2];
#pragma unroll
    for (int t = 0; t < 2; ++t) {
#pragma unroll
        for (int r = 0; r < 8; ++r) sc[t].f[r] = 0.0f;
        const int keyr = EDGE ? min(kb + t * 16 + l16, S_LEN - 1)
                              : (kb + t * 16 + l16); // A row = key
#pragma unroll
        for (int db = 0; db < 2; ++db) {
            FragB16 ak;
#pragma unroll
            for (int j = 0; j < 8; ++j)
                ak.u[j] = *(const uint32_t*)&Kb[(size_t)keyr * DK + db * 32 +
                                                kmap(j, half)];
            sc[t].v = __builtin_amdgcn_wmma_f32_16x16x32_bf16(
                false, ak.v, false, bq[db].v, (short)0, sc[t].v, false, false);
        }
    }

    // ---- spatial bias + mask; lane-local max over this lane's 16 keys ----
    float mb = -3.0e38f;
#pragma unroll
    for (int t = 0; t < 2; ++t)
#pragma unroll
        for (int r = 0; r < 8; ++r) {
            const int key = kb + t * 16 + r + 8 * half;
            float s;
            if (!EDGE || key < S_LEN) {
                int dh = hqv - key / 30; dh = dh < 0 ? -dh : dh;
                int dw = wqv - key % 30; dw = dw < 0 ? -dw : dw;
                s = sc[t].f[r] * 0.125f + s_bias[dh + dw];
            } else {
                s = -1.0e30f;
            }
            sc[t].f[r] = s;
            mb = fmaxf(mb, s);
        }
    mb = fmaxf(mb, __shfl_xor(mb, 16, 32));           // merge key-halves
    const float mnew = fmaxf(m_run, mb);
    const float corr = __expf(m_run - mnew);
    float lb = 0.0f;
#pragma unroll
    for (int t = 0; t < 2; ++t)
#pragma unroll
        for (int r = 0; r < 8; ++r) {
            const float p = __expf(sc[t].f[r] - mnew);
            sc[t].f[r] = p;
            lb += p;
        }
    lb += __shfl_xor(lb, 16, 32);                     // merge key-halves
    l_run = l_run * corr + lb;
    m_run = mnew;

    // rescale ctx^T accumulators by lane-scalar correction
#pragma unroll
    for (int cg = 0; cg < 4; ++cg)
#pragma unroll
        for (int r = 0; r < 8; ++r) cacc[cg].f[r] *= corr;

    // ---- P^T (C layout) -> LDS [q][key] -> B fragment ----
#pragma unroll
    for (int t = 0; t < 2; ++t)
#pragma unroll
        for (int i = 0; i < 4; ++i)
            *(uint32_t*)&pl[l16 * 32 + t * 16 + 8 * half + 2 * i] =
                pack_bf16(sc[t].f[2 * i], sc[t].f[2 * i + 1]);
    asm volatile("s_wait_dscnt 0" ::: "memory");      // same-wave LDS RAW fence
    FragB16 bp;
#pragma unroll
    for (int j = 0; j < 8; ++j)
        bp.u[j] = *(const uint32_t*)&pl[l16 * 32 + kmap(j, half)];

    // ---- ctx^T += V^T_frag @ P^T (4 WMMAs) ----
#pragma unroll
    for (int cg = 0; cg < 4; ++cg)
        cacc[cg].v = __builtin_amdgcn_wmma_f32_16x16x32_bf16(
            false, av[cg].v, false, bp.v, (short)0, cacc[cg].v, false, false);
}

// ---------------------------------------------------------------------------
// Flash-style attention, fully transposed. One wave per 16-query tile per
// (b,h); loops over 32-key blocks.
//   S^T[key, q]  = K_blk(16k x 64d) @ Q^T(64d x 16q)   -> lane l16 = query col
//   ctx^T[d, q] += V^T_frag(16d x 32k) @ P^T(32k x 16q)
// ---------------------------------------------------------------------------
__global__ __launch_bounds__(128) void attn_kernel(
    const uint16_t* __restrict__ Q, const uint16_t* __restrict__ K,
    const uint16_t* __restrict__ VT, const float* __restrict__ sbias,
    float* __restrict__ ctx_out) {
    __shared__ float s_bias[64];
    __shared__ uint16_t p_lds[4][16 * 32];  // per-wave P^T tile [q][key] bf16

    const int tid = threadIdx.x;
    if (tid < 59) s_bias[tid] = sbias[(size_t)tid * 59];  // column 0 of table
    __syncthreads();

    const int wave = tid >> 5, lane = tid & 31;
    const int half = lane >> 4, l16 = lane & 15;
    const int qt = blockIdx.x * 4 + wave;  // query tile 0..56
    const int bh = blockIdx.y;             // b*8+h, 0..127
    if (qt >= 57) return;
    const int q0 = qt * 16;

    const uint16_t* Qb = Q + (size_t)bh * S_LEN * DK;
    const uint16_t* Kb = K + (size_t)bh * S_LEN * DK;
    const uint16_t* Vb = VT + (size_t)bh * DK * S_LEN;
    uint16_t* pl = &p_lds[wave][0];

    // B fragments of Q^T (cols = queries), d-blocks 0..31 / 32..63; loop-invariant
    const int q = q0 + l16;
    const int qc = min(q, S_LEN - 1);
    FragB16 bq[2];
#pragma unroll
    for (int db = 0; db < 2; ++db)
#pragma unroll
        for (int j = 0; j < 8; ++j)
            bq[db].u[j] =
                *(const uint32_t*)&Qb[(size_t)qc * DK + db * 32 + kmap(j, half)];

    const int hqv = qc / 30, wqv = qc % 30;  // this lane's query grid coords

    float m_run = -3.0e38f, l_run = 0.0f;    // lane-scalar online-softmax state
    FragF32 cacc[4];                          // ctx^T accumulators, d blocks of 16
#pragma unroll
    for (int cg = 0; cg < 4; ++cg)
#pragma unroll
        for (int r = 0; r < 8; ++r) cacc[cg].f[r] = 0.0f;

    for (int kb = 0; kb < 896; kb += 32) {   // 28 full blocks, zero guards
        __builtin_prefetch(&Kb[(size_t)(kb + 32) * DK], 0, 1);
        __builtin_prefetch(&Vb[kb + 32], 0, 1);
        attn_block<false>(kb, Kb, Vb, bq, cacc, m_run, l_run, hqv, wqv, half,
                          l16, pl, s_bias);
    }
    attn_block<true>(896, Kb, Vb, bq, cacc, m_run, l_run, hqv, wqv, half, l16,
                     pl, s_bias);            // tail: keys 896..899 valid

    // ---- normalize and store: lane owns query q; d = cg*16 + 8*half + r ----
    if (q < S_LEN) {
        const int b = bh >> 3, h = bh & 7;
        const float inv = 1.0f / l_run;
        float* dst = ctx_out + ((size_t)(b * S_LEN + q)) * D_MODEL + h * DK;
#pragma unroll
        for (int cg = 0; cg < 4; ++cg) {
            const int dbase = cg * 16 + 8 * half;
            float4 v0 = make_float4(cacc[cg].f[0] * inv, cacc[cg].f[1] * inv,
                                    cacc[cg].f[2] * inv, cacc[cg].f[3] * inv);
            float4 v1 = make_float4(cacc[cg].f[4] * inv, cacc[cg].f[5] * inv,
                                    cacc[cg].f[6] * inv, cacc[cg].f[7] * inv);
            ((float4*)(dst + dbase))[0] = v0;
            ((float4*)(dst + dbase))[1] = v1;
        }
    }
}

// ---------------------------------------------------------------------------
// LayerNorm: one wave per 512-element row.
// ---------------------------------------------------------------------------
__global__ __launch_bounds__(128) void ln_kernel(
    const float* __restrict__ resid, const float* __restrict__ gamma,
    const float* __restrict__ beta, float* __restrict__ out) {
    const int wave = threadIdx.x >> 5, lane = threadIdx.x & 31;
    const int row = blockIdx.x * 4 + wave;
    const float* r = resid + (size_t)row * D_MODEL;
    float vals[16];
    float sum = 0.0f, sq = 0.0f;
#pragma unroll
    for (int i = 0; i < 4; ++i) {
        float4 f = ((const float4*)r)[lane + i * 32];
        vals[i * 4 + 0] = f.x; vals[i * 4 + 1] = f.y;
        vals[i * 4 + 2] = f.z; vals[i * 4 + 3] = f.w;
        sum += f.x + f.y + f.z + f.w;
        sq  += f.x * f.x + f.y * f.y + f.z * f.z + f.w * f.w;
    }
#pragma unroll
    for (int m = 16; m; m >>= 1) {
        sum += __shfl_xor(sum, m, 32);
        sq  += __shfl_xor(sq,  m, 32);
    }
    const float mu = sum * (1.0f / 512.0f);
    const float var = sq * (1.0f / 512.0f) - mu * mu;
    const float rs = rsqrtf(var + 1e-5f);
#pragma unroll
    for (int i = 0; i < 4; ++i)
#pragma unroll
        for (int j = 0; j < 4; ++j) {
            const int col = (lane + i * 32) * 4 + j;
            out[(size_t)row * D_MODEL + col] =
                (vals[i * 4 + j] - mu) * rs * gamma[col] + beta[col];
        }
}

// ---------------------------------------------------------------------------
extern "C" void kernel_launch(void* const* d_in, const int* in_sizes, int n_in,
                              void* d_out, int out_size, void* d_ws,
                              size_t ws_size, hipStream_t stream) {
    const float* x     = (const float*)d_in[0];
    const float* wq    = (const float*)d_in[1];
    const float* bq    = (const float*)d_in[2];
    const float* wk    = (const float*)d_in[3];
    const float* bk    = (const float*)d_in[4];
    const float* wv    = (const float*)d_in[5];
    const float* bv    = (const float*)d_in[6];
    const float* wo    = (const float*)d_in[7];
    const float* bo    = (const float*)d_in[8];
    const float* gamma = (const float*)d_in[9];
    const float* beta  = (const float*)d_in[10];
    const float* sbias = (const float*)d_in[11];

    // workspace layout (~103 MB total)
    uint8_t* ws = (uint8_t*)d_ws;
    const size_t SZ_H = (size_t)M_ROWS * D_MODEL * sizeof(uint16_t);  // 14.75 MB
    const size_t SZ_F = (size_t)M_ROWS * D_MODEL * sizeof(float);     // 29.49 MB
    uint16_t* Qw = (uint16_t*)(ws);
    uint16_t* Kw = (uint16_t*)(ws + SZ_H);
    uint16_t* Vt = (uint16_t*)(ws + 2 * SZ_H);
    float* ctx   = (float*)(ws + 3 * SZ_H + 4096);  // pad behind Vt for overread
    float* resid = (float*)(ws + 3 * SZ_H + 4096 + SZ_F);

    const dim3 gG(M_ROWS / 64, D_MODEL / 64);  // 225 x 8
    const dim3 blk(128);
    gemm512<EPI_HEADS><<<gG, blk, 0, stream>>>(x, wq, bq, nullptr, Qw);
    gemm512<EPI_HEADS><<<gG, blk, 0, stream>>>(x, wk, bk, nullptr, Kw);
    gemm512<EPI_VT><<<gG, blk, 0, stream>>>(x, wv, bv, nullptr, Vt);
    attn_kernel<<<dim3(15, 128), blk, 0, stream>>>(Qw, Kw, Vt, sbias, ctx);
    gemm512<EPI_RESID><<<gG, blk, 0, stream>>>(ctx, wo, bo, x, resid);
    ln_kernel<<<dim3(M_ROWS / 4), blk, 0, stream>>>(resid, gamma, beta,
                                                    (float*)d_out);
}
